// Qwen3MoeSparseMoeBlock_90898687852693
// MI455X (gfx1250) — compile-verified, matrix operand order checked by reading, not run on hardware
//
#include <hip/hip_runtime.h>
#include <stdint.h>

#define TOKENS   8192
#define HDIM     2048
#define IDIM     768
#define NEXP     64
#define TOPK     8
#define NROWS    (TOKENS * TOPK)   // 65536 expanded rows
#define LSTRIDE  40                // LDS row stride in u16 (80B: 16B-aligned, conflict-free)

typedef __attribute__((ext_vector_type(16))) __bf16 v16bf;
typedef __attribute__((ext_vector_type(8)))  float  v8f;

union FragBF { v16bf v; uint32_t u[8]; };

__device__ __forceinline__ uint16_t f2bf(float f) {
  uint32_t u = __float_as_uint(f);
  uint32_t r = (u + 0x7FFFu + ((u >> 16) & 1u)) >> 16;   // RNE
  return (uint16_t)r;
}
__device__ __forceinline__ float bf2f(uint16_t b) {
  return __uint_as_float(((uint32_t)b) << 16);
}
__device__ __forceinline__ uint32_t pk2(float lo, float hi) {
  return (uint32_t)f2bf(lo) | ((uint32_t)f2bf(hi) << 16);
}

// Async 16B global -> LDS copy (per active lane), tracked by ASYNCcnt.
__device__ __forceinline__ void cp_async16(const uint16_t* g, const uint16_t* l) {
  asm volatile("global_load_async_to_lds_b128 %0, %1, off"
               :: "v"((uint32_t)(uintptr_t)l), "v"(g)
               : "memory");
}
#define WAIT_ASYNC(n) asm volatile("s_wait_asynccnt " #n ::: "memory")

// One 16x32 bf16 WMMA operand fragment from LDS (rows = M for A, N for B
// stored N-major).  Per ISA 7.12.2: lanes 0-15 hold row=lane, K pairs:
//   v0..3 -> K=2v,2v+1 ; v4..7 -> K=16+2(v-4),+1 ; lanes 16-31: +8.
__device__ __forceinline__ v16bf load_frag(const uint16_t* base, int lane) {
  FragBF f;
  int m = lane & 15;
  int h = lane >> 4;
  const uint16_t* row = base + m * LSTRIDE;
#pragma unroll
  for (int v = 0; v < 8; ++v) {
    int k = ((v < 4) ? (2 * v) : (16 + 2 * (v - 4))) + 8 * h;
    f.u[v] = *(const uint32_t*)(row + k);
  }
  return f.v;
}

// ------------------------------------------------------------ pre-passes ----
// x fp32 -> bf16 (8 elements / thread)
__global__ void __launch_bounds__(256) moe_cvt_x(
    const float* __restrict__ in, uint16_t* __restrict__ out) {
  int i = blockIdx.x * 256 + threadIdx.x;
  const float4* s = (const float4*)in + (size_t)i * 2;
  float4 a = s[0], b = s[1];
  uint4 o;
  o.x = pk2(a.x, a.y); o.y = pk2(a.z, a.w);
  o.z = pk2(b.x, b.y); o.w = pk2(b.z, b.w);
  ((uint4*)out)[i] = o;
}

// Per-expert [K][N] fp32 -> [N][K] bf16 (64x64 tiles through LDS)
__global__ void __launch_bounds__(256) moe_cvt_tr(
    const float* __restrict__ in, uint16_t* __restrict__ out, int K, int N) {
  int e = blockIdx.z;
  const float* src = in + (size_t)e * K * N;
  uint16_t* dst = out + (size_t)e * K * N;
  __shared__ uint16_t tile[64][65];
  int k0 = blockIdx.y * 64, n0 = blockIdx.x * 64;
  int tr = threadIdx.x >> 4;            // 0..15
  int tc = (threadIdx.x & 15) * 4;      // 0..60
#pragma unroll
  for (int i = 0; i < 4; ++i) {
    int k = tr + i * 16;
    float4 v = *(const float4*)(src + (size_t)(k0 + k) * N + n0 + tc);
    tile[k][tc + 0] = f2bf(v.x);
    tile[k][tc + 1] = f2bf(v.y);
    tile[k][tc + 2] = f2bf(v.z);
    tile[k][tc + 3] = f2bf(v.w);
  }
  __syncthreads();
#pragma unroll
  for (int i = 0; i < 4; ++i) {
    int n = tr + i * 16;
    uint32_t a = (uint32_t)tile[tc + 0][n] | ((uint32_t)tile[tc + 1][n] << 16);
    uint32_t b = (uint32_t)tile[tc + 2][n] | ((uint32_t)tile[tc + 3][n] << 16);
    uint32_t* p = (uint32_t*)(dst + (size_t)(n0 + n) * K + k0 + tc);
    p[0] = a; p[1] = b;
  }
}

// ---------------------------------------------------------------- router ----
__global__ void __launch_bounds__(256) moe_router(
    const float* __restrict__ x, const float* __restrict__ gw,
    int* __restrict__ sel, float* __restrict__ wts, unsigned* __restrict__ counts) {
  int lane = threadIdx.x & 31;
  int t = blockIdx.x * 8 + (threadIdx.x >> 5);
  const float* xr = x + (size_t)t * HDIM;
  float a0 = 0.f, a1 = 0.f;              // experts: lane and lane+32
  for (int h = 0; h < HDIM; h += 4) {
    float4 xv = *(const float4*)(xr + h);
    const float* g = gw + (size_t)h * NEXP + lane;
    a0 = fmaf(xv.x, g[0 * NEXP], a0);       a1 = fmaf(xv.x, g[0 * NEXP + 32], a1);
    a0 = fmaf(xv.y, g[1 * NEXP], a0);       a1 = fmaf(xv.y, g[1 * NEXP + 32], a1);
    a0 = fmaf(xv.z, g[2 * NEXP], a0);       a1 = fmaf(xv.z, g[2 * NEXP + 32], a1);
    a0 = fmaf(xv.w, g[3 * NEXP], a0);       a1 = fmaf(xv.w, g[3 * NEXP + 32], a1);
  }
  float m = fmaxf(a0, a1);
  for (int off = 16; off; off >>= 1) m = fmaxf(m, __shfl_xor(m, off, 32));
  float e0 = __expf(a0 - m), e1 = __expf(a1 - m);
  float s = e0 + e1;
  for (int off = 16; off; off >>= 1) s += __shfl_xor(s, off, 32);
  float p0 = e0 / s, p1 = e1 / s;
  float wsel[TOPK]; int esel[TOPK]; float sum8 = 0.f;
#pragma unroll
  for (int it = 0; it < TOPK; ++it) {
    float v; int e;
    if (p0 >= p1) { v = p0; e = lane; } else { v = p1; e = lane + 32; }
    for (int off = 16; off; off >>= 1) {
      float ov = __shfl_xor(v, off, 32);
      int   oe = __shfl_xor(e, off, 32);
      if (ov > v || (ov == v && oe < e)) { v = ov; e = oe; }
    }
    wsel[it] = v; esel[it] = e; sum8 += v;
    if ((e & 31) == lane) { if (e < 32) p0 = -1.f; else p1 = -1.f; }
  }
  if (lane == 0) {
#pragma unroll
    for (int it = 0; it < TOPK; ++it) {
      sel[t * TOPK + it] = esel[it];
      wts[t * TOPK + it] = wsel[it] / sum8;      // norm_topk_prob
      atomicAdd(&counts[esel[it]], 1u);
    }
  }
}

// ---------------------------------------------------- permutation helpers ---
__global__ void moe_zero(unsigned* p, int n) {
  int i = blockIdx.x * blockDim.x + threadIdx.x;
  if (i < n) p[i] = 0u;
}
__global__ void moe_scan(const unsigned* __restrict__ counts, unsigned* __restrict__ base) {
  if (threadIdx.x == 0) {
    unsigned a = 0;
    for (int e = 0; e < NEXP; ++e) { base[e] = a; a += counts[e]; }
  }
}
__global__ void __launch_bounds__(256) moe_build(
    const int* __restrict__ sel, const unsigned* __restrict__ base,
    unsigned* __restrict__ cursors, int* __restrict__ row_token, int* __restrict__ pos_of) {
  int idx = blockIdx.x * 256 + threadIdx.x;
  if (idx >= NROWS) return;
  int e = sel[idx];
  unsigned p = base[e] + atomicAdd(&cursors[e], 1u);
  row_token[p] = idx >> 3;
  pos_of[idx] = (int)p;
}

// ------------------------------------------- fused gate/up grouped GEMM -----
// grid: (IDIM/128, NROWS/128, NEXP), block 256 (8 waves as 4M x 2N)
__global__ void __launch_bounds__(256) moe_gateup(
    const uint16_t* __restrict__ xbf, const uint16_t* __restrict__ gT,
    const uint16_t* __restrict__ uT, const int* __restrict__ row_token,
    const unsigned* __restrict__ counts, const unsigned* __restrict__ base,
    uint16_t* __restrict__ inter) {
  int e = blockIdx.z;
  int rows = (int)counts[e];
  int m0 = blockIdx.y * 128;
  if (m0 >= rows) return;                       // dead tile, early exit
  int n0 = blockIdx.x * 128;
  unsigned pos0 = base[e];

  __shared__ __align__(16) uint16_t lsA [2][128 * LSTRIDE];   // [m][k]
  __shared__ __align__(16) uint16_t lsBg[2][128 * LSTRIDE];   // [n][k]
  __shared__ __align__(16) uint16_t lsBu[2][128 * LSTRIDE];

  int tid = threadIdx.x;
  int lane = tid & 31, w = tid >> 5;
  int wm = w & 3, wn = w >> 2;

  // copy decomposition: 2 threads/row, 16 bf16 (two b128) per thread
  int r = tid >> 1, h16 = (tid & 1) * 16;
  int arow = m0 + r;
  int atok = row_token[pos0 + (arow < rows ? arow : 0)];
  const uint16_t* xrow = xbf + (size_t)atok * HDIM + h16;
  const uint16_t* grow = gT + ((size_t)e * IDIM + n0 + r) * HDIM + h16;
  const uint16_t* urow = uT + ((size_t)e * IDIM + n0 + r) * HDIM + h16;
  int lo = r * LSTRIDE + h16;

  auto issue = [&](int buf, int kk) {            // 6 async b128 per thread
    cp_async16(xrow + kk,     &lsA [buf][lo]);
    cp_async16(xrow + kk + 8, &lsA [buf][lo + 8]);
    cp_async16(grow + kk,     &lsBg[buf][lo]);
    cp_async16(grow + kk + 8, &lsBg[buf][lo + 8]);
    cp_async16(urow + kk,     &lsBu[buf][lo]);
    cp_async16(urow + kk + 8, &lsBu[buf][lo + 8]);
  };

  const v8f vzero = {0.f, 0.f, 0.f, 0.f, 0.f, 0.f, 0.f, 0.f};
  v8f accg[2][4], accu[2][4];
#pragma unroll
  for (int i = 0; i < 2; ++i)
#pragma unroll
    for (int j = 0; j < 4; ++j) { accg[i][j] = vzero; accu[i][j] = vzero; }

  issue(0, 0);
  for (int s = 0, kk = 0; kk < HDIM; ++s, kk += 32) {
    int cur = s & 1;
    if (kk + 32 < HDIM) {
      issue(cur ^ 1, kk + 32);
      WAIT_ASYNC(6);            // current buffer's 6 copies done, next in flight
    } else {
      WAIT_ASYNC(0);
    }
    __syncthreads();

    v16bf a0f = load_frag(&lsA[cur][(wm * 32 + 0)  * LSTRIDE], lane);
    v16bf a1f = load_frag(&lsA[cur][(wm * 32 + 16) * LSTRIDE], lane);
#pragma unroll
    for (int nf = 0; nf < 4; ++nf) {
      v16bf bg = load_frag(&lsBg[cur][(wn * 64 + nf * 16) * LSTRIDE], lane);
      accg[0][nf] = __builtin_amdgcn_wmma_f32_16x16x32_bf16(false, a0f, false, bg, (short)0, accg[0][nf], false, false);
      accg[1][nf] = __builtin_amdgcn_wmma_f32_16x16x32_bf16(false, a1f, false, bg, (short)0, accg[1][nf], false, false);
      v16bf bu = load_frag(&lsBu[cur][(wn * 64 + nf * 16) * LSTRIDE], lane);
      accu[0][nf] = __builtin_amdgcn_wmma_f32_16x16x32_bf16(false, a0f, false, bu, (short)0, accu[0][nf], false, false);
      accu[1][nf] = __builtin_amdgcn_wmma_f32_16x16x32_bf16(false, a1f, false, bu, (short)0, accu[1][nf], false, false);
    }
    __syncthreads();            // all reads of 'cur' done before it is refilled
  }

  // epilogue: silu(gate) * up -> bf16 intermediate
  int ln = lane & 15, lh = lane >> 4;
#pragma unroll
  for (int mf = 0; mf < 2; ++mf)
#pragma unroll
    for (int nf = 0; nf < 4; ++nf)
#pragma unroll
      for (int rr = 0; rr < 8; ++rr) {
        int row = m0 + wm * 32 + mf * 16 + lh * 8 + rr;
        if (row < rows) {
          int col = n0 + wn * 64 + nf * 16 + ln;
          float g = accg[mf][nf][rr];
          float u = accu[mf][nf][rr];
          float sg = g / (1.f + __expf(-g));
          inter[(size_t)(pos0 + row) * IDIM + col] = f2bf(sg * u);
        }
      }
}

// ------------------------------------------------- down-proj grouped GEMM ---
// grid: (HDIM/128, NROWS/128, NEXP), block 256
__global__ void __launch_bounds__(256) moe_down(
    const uint16_t* __restrict__ inter, const uint16_t* __restrict__ dT,
    const unsigned* __restrict__ counts, const unsigned* __restrict__ base,
    uint16_t* __restrict__ eout) {
  int e = blockIdx.z;
  int rows = (int)counts[e];
  int m0 = blockIdx.y * 128;
  if (m0 >= rows) return;
  int n0 = blockIdx.x * 128;
  unsigned pos0 = base[e];

  __shared__ __align__(16) uint16_t lsA[2][128 * LSTRIDE];
  __shared__ __align__(16) uint16_t lsB[2][128 * LSTRIDE];

  int tid = threadIdx.x;
  int lane = tid & 31, w = tid >> 5;
  int wm = w & 3, wn = w >> 2;

  int r = tid >> 1, h16 = (tid & 1) * 16;
  int arow = m0 + r;
  int apos = (int)pos0 + (arow < rows ? arow : 0);
  const uint16_t* irow = inter + (size_t)apos * IDIM + h16;
  const uint16_t* drow = dT + ((size_t)e * HDIM + n0 + r) * IDIM + h16;
  int lo = r * LSTRIDE + h16;

  auto issue = [&](int buf, int kk) {            // 4 async b128 per thread
    cp_async16(irow + kk,     &lsA[buf][lo]);
    cp_async16(irow + kk + 8, &lsA[buf][lo + 8]);
    cp_async16(drow + kk,     &lsB[buf][lo]);
    cp_async16(drow + kk + 8, &lsB[buf][lo + 8]);
  };

  const v8f vzero = {0.f, 0.f, 0.f, 0.f, 0.f, 0.f, 0.f, 0.f};
  v8f acc[2][4];
#pragma unroll
  for (int i = 0; i < 2; ++i)
#pragma unroll
    for (int j = 0; j < 4; ++j) acc[i][j] = vzero;

  issue(0, 0);
  for (int s = 0, kk = 0; kk < IDIM; ++s, kk += 32) {
    int cur = s & 1;
    if (kk + 32 < IDIM) {
      issue(cur ^ 1, kk + 32);
      WAIT_ASYNC(4);
    } else {
      WAIT_ASYNC(0);
    }
    __syncthreads();

    v16bf a0f = load_frag(&lsA[cur][(wm * 32 + 0)  * LSTRIDE], lane);
    v16bf a1f = load_frag(&lsA[cur][(wm * 32 + 16) * LSTRIDE], lane);
#pragma unroll
    for (int nf = 0; nf < 4; ++nf) {
      v16bf bd = load_frag(&lsB[cur][(wn * 64 + nf * 16) * LSTRIDE], lane);
      acc[0][nf] = __builtin_amdgcn_wmma_f32_16x16x32_bf16(false, a0f, false, bd, (short)0, acc[0][nf], false, false);
      acc[1][nf] = __builtin_amdgcn_wmma_f32_16x16x32_bf16(false, a1f, false, bd, (short)0, acc[1][nf], false, false);
    }
    __syncthreads();
  }

  int ln = lane & 15, lh = lane >> 4;
#pragma unroll
  for (int mf = 0; mf < 2; ++mf)
#pragma unroll
    for (int nf = 0; nf < 4; ++nf)
#pragma unroll
      for (int rr = 0; rr < 8; ++rr) {
        int row = m0 + wm * 32 + mf * 16 + lh * 8 + rr;
        if (row < rows) {
          int col = n0 + wn * 64 + nf * 16 + ln;
          eout[(size_t)(pos0 + row) * HDIM + col] = f2bf(acc[mf][nf][rr]);
        }
      }
}

// --------------------------------------------------------------- combine ----
__global__ void __launch_bounds__(256) moe_combine(
    const uint16_t* __restrict__ eout, const float* __restrict__ wts,
    const int* __restrict__ pos_of, uint16_t* __restrict__ out) {
  int t = blockIdx.x;
  float w[TOPK]; int p[TOPK];
#pragma unroll
  for (int k = 0; k < TOPK; ++k) { w[k] = wts[t * TOPK + k]; p[k] = pos_of[t * TOPK + k]; }
  for (int h = threadIdx.x; h < HDIM; h += 256) {
    float acc = 0.f;
#pragma unroll
    for (int k = 0; k < TOPK; ++k)
      acc = fmaf(w[k], bf2f(eout[(size_t)p[k] * HDIM + h]), acc);
    out[(size_t)t * HDIM + h] = f2bf(acc);
  }
}

// ---------------------------------------------------------------- launch ----
extern "C" void kernel_launch(void* const* d_in, const int* in_sizes, int n_in,
                              void* d_out, int out_size, void* d_ws, size_t ws_size,
                              hipStream_t stream) {
  (void)in_sizes; (void)n_in; (void)out_size; (void)ws_size;
  const float* x  = (const float*)d_in[0];
  const float* gw = (const float*)d_in[1];
  const float* gp = (const float*)d_in[2];
  const float* up = (const float*)d_in[3];
  const float* dp = (const float*)d_in[4];

  char* ws = (char*)d_ws;
  size_t off = 0;
  auto take = [&](size_t bytes) {
    size_t o = off;
    off += (bytes + 255) & ~(size_t)255;
    return o;
  };
  unsigned* counts  = (unsigned*)(ws + take(NEXP * 4));
  unsigned* cursors = (unsigned*)(ws + take(NEXP * 4));
  unsigned* base    = (unsigned*)(ws + take(NEXP * 4));
  int*      sel     = (int*)     (ws + take((size_t)NROWS * 4));
  float*    wts     = (float*)   (ws + take((size_t)NROWS * 4));
  int*      row_tok = (int*)     (ws + take((size_t)NROWS * 4));
  int*      pos_of  = (int*)     (ws + take((size_t)NROWS * 4));
  uint16_t* xbf     = (uint16_t*)(ws + take((size_t)TOKENS * HDIM * 2));
  uint16_t* gT      = (uint16_t*)(ws + take((size_t)NEXP * HDIM * IDIM * 2));
  uint16_t* uT      = (uint16_t*)(ws + take((size_t)NEXP * HDIM * IDIM * 2));
  uint16_t* dT      = (uint16_t*)(ws + take((size_t)NEXP * HDIM * IDIM * 2));
  uint16_t* inter   = (uint16_t*)(ws + take((size_t)NROWS * IDIM * 2));
  uint16_t* eout    = (uint16_t*)(ws + take((size_t)NROWS * HDIM * 2));

  moe_zero<<<1, 128, 0, stream>>>(counts, 128);   // counts + cursors (contiguous)
  moe_cvt_x<<<(TOKENS * HDIM / 8) / 256, 256, 0, stream>>>(x, xbf);
  moe_cvt_tr<<<dim3(IDIM / 64, HDIM / 64, NEXP), 256, 0, stream>>>(gp, gT, HDIM, IDIM);
  moe_cvt_tr<<<dim3(IDIM / 64, HDIM / 64, NEXP), 256, 0, stream>>>(up, uT, HDIM, IDIM);
  moe_cvt_tr<<<dim3(HDIM / 64, IDIM / 64, NEXP), 256, 0, stream>>>(dp, dT, IDIM, HDIM);
  moe_router<<<TOKENS / 8, 256, 0, stream>>>(x, gw, sel, wts, counts);
  moe_scan<<<1, 64, 0, stream>>>(counts, base);
  moe_build<<<NROWS / 256, 256, 0, stream>>>(sel, base, cursors, row_tok, pos_of);
  moe_gateup<<<dim3(IDIM / 128, NROWS / 128, NEXP), 256, 0, stream>>>(
      xbf, gT, uT, row_tok, counts, base, inter);
  moe_down<<<dim3(HDIM / 128, NROWS / 128, NEXP), 256, 0, stream>>>(
      inter, dT, counts, base, eout);
  moe_combine<<<TOKENS, 256, 0, stream>>>(eout, wts, pos_of, (uint16_t*)d_out);
}
